// MultiHeadAttention_44195213476062
// MI455X (gfx1250) — matrix operand order repaired and checked by reading, not compiled, not run on hardware
//
#include <hip/hip_runtime.h>

// ---------------------------------------------------------------------------
// MHA forward for MI455X (gfx1250, wave32, WMMA, async global->LDS pipelines).
//   B=2, T=2048, C=1024, H=16, D=64.
// GEMMs: 8-wave block -> 128x64 tile, double-buffered 32-wide K slabs via
// global_load_async_to_lds_b128. Attention: block-cooperative flash attention,
// one block per (b,h,128-q band); K/V slabs shared through LDS (async),
// online softmax per 16-row wave tile, V stored transposed.
// ---------------------------------------------------------------------------

#define EMB 1024
#define NH  16
#define HD  64
#define TT  2048
#define NB  2
#define BT  4096   // NB*TT

#define APAD 40            // padded LDS row stride for 32-wide GEMM slabs
#define ATILE_E (128 * APAD)
#define BTILE_E (64 * APAD)
#define STAGE_E (ATILE_E + BTILE_E)

#define KSTRIDE 72         // K slab: 32 rows x 64 elems (padded)
#define VSTRIDE 40         // V^T slab: 64 rows x 32 elems (padded)

typedef __attribute__((ext_vector_type(16))) __bf16 v16bf;
typedef __attribute__((ext_vector_type(8)))  float  v8f;

#define WMMA_BF16(A, B, C) \
  __builtin_amdgcn_wmma_f32_16x16x32_bf16(false, (A), false, (B), (short)0, (C), false, false)

__device__ __forceinline__ unsigned short f2bf(float f) {
  unsigned int u = __float_as_uint(f);
  u += 0x7fffu + ((u >> 16) & 1u);          // round-to-nearest-even
  return (unsigned short)(u >> 16);
}

// 16 contiguous bf16 (32B): B-fragment for lane L = column N=L&15, K run of 16.
__device__ __forceinline__ v16bf ldB16(const unsigned short* p) {
  union { uint4 q[2]; v16bf v; } u;
  u.q[0] = *(const uint4*)(p);
  u.q[1] = *(const uint4*)(p + 8);
  return u.v;
}
// A-fragment: two 8-element (16B) chunks (K = 16*half + j , K = 16*half + 8 + j).
__device__ __forceinline__ v16bf ldA2(const unsigned short* c0, const unsigned short* c1) {
  union { uint4 q[2]; v16bf v; } u;
  u.q[0] = *(const uint4*)(c0);
  u.q[1] = *(const uint4*)(c1);
  return u.v;
}

// Async 16B global -> LDS copy (per-lane addresses), tracked by ASYNCcnt.
__device__ __forceinline__ void async_cp16(unsigned lds_off, const unsigned short* g) {
  asm volatile("global_load_async_to_lds_b128 %0, %1, off"
               :: "v"(lds_off), "v"(g)
               : "memory");
}
#define WAIT_ASYNC3() asm volatile("s_wait_asynccnt 0x3" ::: "memory")
#define WAIT_ASYNC2() asm volatile("s_wait_asynccnt 0x2" ::: "memory")
#define WAIT_ASYNC0() asm volatile("s_wait_asynccnt 0x0" ::: "memory")

// ---------------------------------------------------------------------------
__global__ __launch_bounds__(256) void cvt_bf16v(const float4* __restrict__ src,
                                                 uint2* __restrict__ dst, int n4) {
  int i = blockIdx.x * blockDim.x + threadIdx.x;
  int stride = gridDim.x * blockDim.x;
  for (; i < n4; i += stride) {
    float4 f = src[i];
    uint2 o;
    o.x = (unsigned)f2bf(f.x) | ((unsigned)f2bf(f.y) << 16);
    o.y = (unsigned)f2bf(f.z) | ((unsigned)f2bf(f.w) << 16);
    dst[i] = o;
  }
}

// ---------------------------------------------------------------------------
// Cooperative GEMM stage load: A slab 128x32 (2 chunks/thread) + B slab 64x32
// (1 chunk/thread) -> 3 async instructions per thread per stage.
__device__ __forceinline__ void stage_load(const unsigned short* __restrict__ Ag,
                                           const unsigned short* __restrict__ Bg,
                                           unsigned aLds, unsigned bLds, int t) {
  const int r0 = t >> 2, s0 = t & 3;
  const int r1 = r0 + 64;
  async_cp16(aLds + (unsigned)(r0 * APAD + s0 * 8) * 2, Ag + (size_t)r0 * EMB + s0 * 8);
  async_cp16(aLds + (unsigned)(r1 * APAD + s0 * 8) * 2, Ag + (size_t)r1 * EMB + s0 * 8);
  async_cp16(bLds + (unsigned)(r0 * APAD + s0 * 8) * 2, Bg + (size_t)r0 * EMB + s0 * 8);
}

// One GEMM compute stage: all LDS fragments first, then 4 back-to-back WMMAs.
__device__ __forceinline__ void gemm_stage(const unsigned short* As, const unsigned short* Bs,
                                           int wv, int lr, int hf, v8f acc[4]) {
  const unsigned short* aRow = As + (wv * 16 + lr) * APAD;
  v16bf a  = ldA2(aRow + 16 * hf, aRow + 16 * hf + 8);
  v16bf b0 = ldB16(Bs + (0 * 16 + lr) * APAD + 16 * hf);
  v16bf b1 = ldB16(Bs + (1 * 16 + lr) * APAD + 16 * hf);
  v16bf b2 = ldB16(Bs + (2 * 16 + lr) * APAD + 16 * hf);
  v16bf b3 = ldB16(Bs + (3 * 16 + lr) * APAD + 16 * hf);
  acc[0] = WMMA_BF16(a, b0, acc[0]);
  acc[1] = WMMA_BF16(a, b1, acc[1]);
  acc[2] = WMMA_BF16(a, b2, acc[2]);
  acc[3] = WMMA_BF16(a, b3, acc[3]);
}

// ---------------------------------------------------------------------------
// qkv = x @ W_qkv^T ; split into Q (scaled 1/8), K [b,h,t,d], V^T [b,h,d,t].
__global__ __launch_bounds__(256) void qkv_gemm(const unsigned short* __restrict__ xb,
                                                const unsigned short* __restrict__ wb,
                                                unsigned short* __restrict__ Qb,
                                                unsigned short* __restrict__ Kb,
                                                unsigned short* __restrict__ Vt) {
  __shared__ unsigned short smem[2][STAGE_E];
  const int t = threadIdx.x;
  const int lane = t & 31, wv = t >> 5;
  const int mB = blockIdx.x / 48, nB = blockIdx.x % 48;   // 32 x 48 blocks
  const int m0 = mB << 7, n0 = nB << 6;
  const int lr = lane & 15, hf = lane >> 4;

  const unsigned aL[2] = { (unsigned)(size_t)&smem[0][0], (unsigned)(size_t)&smem[1][0] };
  const unsigned bL[2] = { (unsigned)(size_t)&smem[0][ATILE_E], (unsigned)(size_t)&smem[1][ATILE_E] };
  const unsigned short* Ag = xb + (size_t)m0 * EMB;
  const unsigned short* Bg = wb + (size_t)n0 * EMB;

  v8f acc[4];
#pragma unroll
  for (int j = 0; j < 4; ++j) acc[j] = v8f{0.f,0.f,0.f,0.f,0.f,0.f,0.f,0.f};

  stage_load(Ag, Bg, aL[0], bL[0], t);
  int s = 0;
  for (; s < EMB / 32 - 1; ++s) {                 // steady state: single-path body
    stage_load(Ag + (s + 1) * 32, Bg + (s + 1) * 32, aL[(s + 1) & 1], bL[(s + 1) & 1], t);
    WAIT_ASYNC3();
    __syncthreads();
    gemm_stage(&smem[s & 1][0], &smem[s & 1][ATILE_E], wv, lr, hf, acc);
    __syncthreads();
  }
  WAIT_ASYNC0();                                  // drain last stage
  __syncthreads();
  gemm_stage(&smem[s & 1][0], &smem[s & 1][ATILE_E], wv, lr, hf, acc);

  const int mw = m0 + wv * 16;
#pragma unroll
  for (int j = 0; j < 4; ++j) {
    const int n = n0 + j * 16 + lr;
    const int sseg = n >> 10, hh = (n >> 6) & 15, d = n & 63;
#pragma unroll
    for (int r = 0; r < 8; ++r) {
      const int m = mw + r + 8 * hf;
      const int bb = m >> 11, tt = m & 2047;
      const float v = acc[j][r];
      if (sseg == 0)      Qb[(size_t)(((bb * NH + hh) << 11) + tt) * HD + d] = f2bf(v * 0.125f);
      else if (sseg == 1) Kb[(size_t)(((bb * NH + hh) << 11) + tt) * HD + d] = f2bf(v);
      else                Vt[(size_t)(((bb * NH + hh) << 6) + d) * TT + tt]  = f2bf(v);
    }
  }
}

// ---------------------------------------------------------------------------
// Block-cooperative flash attention: block = (b, h, 128-row q band), 8 waves.
// K (32x64) and V^T (64x32) slabs double-buffered in LDS via async copies.
__global__ __launch_bounds__(256) void attn_fused(const unsigned short* __restrict__ Qb,
                                                  const unsigned short* __restrict__ Kb,
                                                  const unsigned short* __restrict__ Vt,
                                                  unsigned short* __restrict__ Ob) {
  __shared__ unsigned short kslab[2][32 * KSTRIDE];   // 2 x 4.5 KB
  __shared__ unsigned short vslab[2][64 * VSTRIDE];   // 2 x 5.0 KB
  __shared__ unsigned short pl[8 * 16 * 32];          // per-wave P tiles, 8 KB
  const int t = threadIdx.x;
  const int lane = t & 31, wv = t >> 5;
  unsigned short* pbuf = pl + wv * (16 * 32);
  const int Q0 = (blockIdx.x & 15) << 7;              // q band start
  const int h = (blockIdx.x >> 4) & 15, b = blockIdx.x >> 8;
  const int q0 = Q0 + wv * 16;
  const int lr = lane & 15, hf = lane >> 4;

  const unsigned kL[2] = { (unsigned)(size_t)&kslab[0][0], (unsigned)(size_t)&kslab[1][0] };
  const unsigned vL[2] = { (unsigned)(size_t)&vslab[0][0], (unsigned)(size_t)&vslab[1][0] };

  const size_t bh = (size_t)(b * NH + h);
  const unsigned short* Qbase = Qb + bh * TT * HD;
  const unsigned short* Kbase = Kb + bh * TT * HD;
  const unsigned short* Vbase = Vt + bh * HD * TT;

  const unsigned short* qRow = Qbase + (size_t)(q0 + lr) * HD;
  const v16bf aQ0 = ldA2(qRow + 16 * hf,      qRow + 16 * hf + 8);   // d 0..31
  const v16bf aQ1 = ldA2(qRow + 32 + 16 * hf, qRow + 40 + 16 * hf);   // d 32..63

  v8f out[4];
  float mi[8], li[8], al[8];
#pragma unroll
  for (int j = 0; j < 4; ++j) out[j] = v8f{0.f,0.f,0.f,0.f,0.f,0.f,0.f,0.f};
#pragma unroll
  for (int r = 0; r < 8; ++r) { mi[r] = -1e30f; li[r] = 0.f; }

  // Cooperative K/V stage: 1 K chunk + 1 V chunk per thread (16B each).
  const int krow = t >> 3, ksub = t & 7;              // 32 rows x 8 chunks
  const int vrow = t >> 2, vsub = t & 3;              // 64 rows x 4 chunks
  const unsigned kOffL = (unsigned)(krow * KSTRIDE + ksub * 8) * 2;
  const unsigned vOffL = (unsigned)(vrow * VSTRIDE + vsub * 8) * 2;

#define ATTN_ISSUE(kk0, buf)                                                        \
  do {                                                                              \
    async_cp16(kL[buf] + kOffL, Kbase + (size_t)((kk0) + krow) * HD + ksub * 8);    \
    async_cp16(vL[buf] + vOffL, Vbase + (size_t)vrow * TT + (kk0) + vsub * 8);      \
  } while (0)

  const int nStages = (Q0 + 128) / 32;                // uniform across the block

#define ATTN_COMPUTE(stageIdx, buf)                                                 \
  do {                                                                              \
    const int kk0 = (stageIdx) * 32;                                                \
    if (kk0 < q0 + 16) {                                                            \
      const unsigned short* Ks = &kslab[buf][0];                                    \
      const unsigned short* Vs = &vslab[buf][0];                                    \
      v8f sA = v8f{0.f,0.f,0.f,0.f,0.f,0.f,0.f,0.f};                                \
      v8f sB = v8f{0.f,0.f,0.f,0.f,0.f,0.f,0.f,0.f};                                \
      const unsigned short* kA = Ks + lr * KSTRIDE + 16 * hf;                       \
      const unsigned short* kB = Ks + (16 + lr) * KSTRIDE + 16 * hf;                \
      v16bf bk0 = ldB16(kA), bk1 = ldB16(kA + 32);                                  \
      v16bf bk2 = ldB16(kB), bk3 = ldB16(kB + 32);                                  \
      v16bf bV0 = ldB16(Vs + (0 * 16 + lr) * VSTRIDE + 16 * hf);                    \
      v16bf bV1 = ldB16(Vs + (1 * 16 + lr) * VSTRIDE + 16 * hf);                    \
      v16bf bV2 = ldB16(Vs + (2 * 16 + lr) * VSTRIDE + 16 * hf);                    \
      v16bf bV3 = ldB16(Vs + (3 * 16 + lr) * VSTRIDE + 16 * hf);                    \
      sA = WMMA_BF16(aQ0, bk0, sA);                                                 \
      sA = WMMA_BF16(aQ1, bk1, sA);                                                 \
      sB = WMMA_BF16(aQ0, bk2, sB);                                                 \
      sB = WMMA_BF16(aQ1, bk3, sB);                                                 \
      _Pragma("unroll")                                                             \
      for (int r = 0; r < 8; ++r) {                                                 \
        const int q = q0 + r + 8 * hf;                                              \
        const float vA = (kk0 + lr      > q) ? -1e30f : sA[r];                      \
        const float vB = (kk0 + 16 + lr > q) ? -1e30f : sB[r];                      \
        float mx = fmaxf(vA, vB);                                                   \
        _Pragma("unroll")                                                           \
        for (int msk = 1; msk < 16; msk <<= 1) mx = fmaxf(mx, __shfl_xor(mx, msk, 32)); \
        const float mn = fmaxf(mi[r], mx);                                          \
        al[r] = __expf(mi[r] - mn);                                                 \
        const float pA = __expf(vA - mn);                                           \
        const float pB = __expf(vB - mn);                                           \
        float ls = pA + pB;                                                         \
        _Pragma("unroll")                                                           \
        for (int msk = 1; msk < 16; msk <<= 1) ls += __shfl_xor(ls, msk, 32);       \
        li[r] = li[r] * al[r] + ls;                                                 \
        mi[r] = mn;                                                                 \
        pbuf[(r + 8 * hf) * 32 + lr]      = f2bf(pA);                               \
        pbuf[(r + 8 * hf) * 32 + 16 + lr] = f2bf(pB);                               \
      }                                                                             \
      _Pragma("unroll")                                                             \
      for (int j = 0; j < 4; ++j)                                                   \
        _Pragma("unroll")                                                           \
        for (int r = 0; r < 8; ++r) out[j][r] *= al[r];                             \
      asm volatile("s_wait_dscnt 0x0" ::: "memory");                                \
      const v16bf pF = ldA2(pbuf + lr * 32 + 16 * hf, pbuf + lr * 32 + 16 * hf + 8); \
      out[0] = WMMA_BF16(pF, bV0, out[0]);                                          \
      out[1] = WMMA_BF16(pF, bV1, out[1]);                                          \
      out[2] = WMMA_BF16(pF, bV2, out[2]);                                          \
      out[3] = WMMA_BF16(pF, bV3, out[3]);                                          \
    }                                                                               \
  } while (0)

  ATTN_ISSUE(0, 0);
  int s = 0;
  for (; s < nStages - 1; ++s) {
    ATTN_ISSUE((s + 1) * 32, (s + 1) & 1);
    WAIT_ASYNC2();
    __syncthreads();
    ATTN_COMPUTE(s, s & 1);
    __syncthreads();
  }
  WAIT_ASYNC0();
  __syncthreads();
  ATTN_COMPUTE(s, s & 1);

  // ---- epilogue: divide by l, write attention output bf16 [b, t, h*64+d]
#pragma unroll
  for (int j = 0; j < 4; ++j) {
#pragma unroll
    for (int r = 0; r < 8; ++r) {
      const int tt = q0 + r + 8 * hf;
      const float v = out[j][r] / li[r];
      Ob[(size_t)(b * TT + tt) * EMB + h * HD + j * 16 + lr] = f2bf(v);
    }
  }
#undef ATTN_ISSUE
#undef ATTN_COMPUTE
}

// ---------------------------------------------------------------------------
// out = attn_out @ W_proj^T.  Block tile 128x64, same async-LDS pipeline.
__global__ __launch_bounds__(256) void proj_gemm(const unsigned short* __restrict__ Ab,
                                                 const unsigned short* __restrict__ wb,
                                                 float* __restrict__ outp) {
  __shared__ unsigned short smem[2][STAGE_E];
  const int t = threadIdx.x;
  const int lane = t & 31, wv = t >> 5;
  const int mB = blockIdx.x >> 4, nB = blockIdx.x & 15;   // 32 x 16 blocks
  const int m0 = mB << 7, n0 = nB << 6;
  const int lr = lane & 15, hf = lane >> 4;

  const unsigned aL[2] = { (unsigned)(size_t)&smem[0][0], (unsigned)(size_t)&smem[1][0] };
  const unsigned bL[2] = { (unsigned)(size_t)&smem[0][ATILE_E], (unsigned)(size_t)&smem[1][ATILE_E] };
  const unsigned short* Ag = Ab + (size_t)m0 * EMB;
  const unsigned short* Bg = wb + (size_t)n0 * EMB;

  v8f acc[4];
#pragma unroll
  for (int j = 0; j < 4; ++j) acc[j] = v8f{0.f,0.f,0.f,0.f,0.f,0.f,0.f,0.f};

  stage_load(Ag, Bg, aL[0], bL[0], t);
  int s = 0;
  for (; s < EMB / 32 - 1; ++s) {
    stage_load(Ag + (s + 1) * 32, Bg + (s + 1) * 32, aL[(s + 1) & 1], bL[(s + 1) & 1], t);
    WAIT_ASYNC3();
    __syncthreads();
    gemm_stage(&smem[s & 1][0], &smem[s & 1][ATILE_E], wv, lr, hf, acc);
    __syncthreads();
  }
  WAIT_ASYNC0();
  __syncthreads();
  gemm_stage(&smem[s & 1][0], &smem[s & 1][ATILE_E], wv, lr, hf, acc);

  const int mw = m0 + wv * 16;
#pragma unroll
  for (int j = 0; j < 4; ++j)
#pragma unroll
    for (int r = 0; r < 8; ++r)
      outp[(size_t)(mw + r + 8 * hf) * EMB + n0 + j * 16 + lr] = acc[j][r];
}

// ---------------------------------------------------------------------------
extern "C" void kernel_launch(void* const* d_in, const int* in_sizes, int n_in,
                              void* d_out, int out_size, void* d_ws, size_t ws_size,
                              hipStream_t stream) {
  const float* x     = (const float*)d_in[0];
  const float* Wqkv  = (const float*)d_in[1];
  const float* Wproj = (const float*)d_in[2];
  float* outp = (float*)d_out;

  // Workspace (bf16 copies + Q/K/Vt/attn-out), ~48 MB total.
  unsigned short* xb     = (unsigned short*)d_ws;
  unsigned short* wqkvb  = xb     + (size_t)BT * EMB;
  unsigned short* wprojb = wqkvb  + (size_t)3 * EMB * EMB;
  unsigned short* Qb     = wprojb + (size_t)EMB * EMB;
  unsigned short* Kb     = Qb     + (size_t)BT * EMB;
  unsigned short* Vt     = Kb     + (size_t)BT * EMB;
  unsigned short* Ob     = Vt     + (size_t)BT * EMB;

  cvt_bf16v<<<512, 256, 0, stream>>>((const float4*)x,     (uint2*)xb,     BT * EMB / 4);
  cvt_bf16v<<<512, 256, 0, stream>>>((const float4*)Wqkv,  (uint2*)wqkvb,  3 * EMB * EMB / 4);
  cvt_bf16v<<<256, 256, 0, stream>>>((const float4*)Wproj, (uint2*)wprojb, EMB * EMB / 4);

  qkv_gemm<<<32 * 48, 256, 0, stream>>>(xb, wqkvb, Qb, Kb, Vt);
  attn_fused<<<512, 256, 0, stream>>>(Qb, Kb, Vt, Ob);   // (T/128)*H*B = 512 blocks
  proj_gemm<<<32 * 16, 256, 0, stream>>>(Ob, wprojb, outp);
}
